// AdaptiveSemiseparableLayer_4475355922623
// MI455X (gfx1250) — matrix operand order repaired and checked by reading, not compiled, not run on hardware
//
#include <hip/hip_runtime.h>
#include <stdint.h>

// ---------------------------------------------------------------------------
// AdaptiveSemiseparableLayer for MI455X (gfx1250), bf16 WMMA pipeline with
// double-buffered async global->LDS DMA (ASYNCcnt software pipeline).
// B=4, L=2048, D=2048, R=2048, DH=512, M=B*L=8192.
// ---------------------------------------------------------------------------

typedef __bf16 bf16_t;
typedef __attribute__((ext_vector_type(16))) bf16_t v16bf;
typedef __attribute__((ext_vector_type(8)))  float  v8f;
typedef __attribute__((ext_vector_type(4)))  unsigned int u32x4;

union FragAB { u32x4 q[2]; v16bf m; };

__device__ __forceinline__ unsigned short f2bf(float f) {
  unsigned int u = __float_as_uint(f);
  u += 0x7FFFu + ((u >> 16) & 1u);        // round-to-nearest-even
  return (unsigned short)(u >> 16);
}

__device__ __forceinline__ void async_b128(void* lds_dst, const void* gsrc) {
  unsigned lds = (unsigned)(uintptr_t)lds_dst;     // low 32b of generic == LDS offset
  asm volatile("global_load_async_to_lds_b128 %0, %1, off"
               :: "v"(lds), "v"(gsrc) : "memory");
}
template <int N>
__device__ __forceinline__ void wait_async_le() {
  asm volatile("s_wait_asynccnt %0" :: "n"(N) : "memory");
}

// -------------------- f32 -> bf16 conversion --------------------
__global__ void cvt_f32_bf16(const float* __restrict__ in,
                             unsigned short* __restrict__ out, long long n) {
  long long i = (long long)blockIdx.x * blockDim.x + threadIdx.x;
  long long stride = (long long)gridDim.x * blockDim.x;
  for (; i < n; i += stride) out[i] = f2bf(in[i]);
}

// -------------------- tiled bf16 WMMA GEMM --------------------
// C(M,N) = A(M,K) * B(K,N), B-element (k,n) read from Bw[n*ldb + k].
// Tile: 128x128 per workgroup (256 thr = 8 waves), wave = 64x32 (4x2 WMMA tiles).
// Double-buffered LDS stages of BK=32; stage s+1 DMA overlaps stage s WMMAs.
// MODE 0: relu  -> bf16 store (h)
// MODE 1: sigmoid -> f32 store (gates)
// MODE 2: (+bias)*gates -> f32 store (u_gated / v_gated)
// MODE 3: + depthwise conv(x) -> f32 store (final output)
#define BM 128
#define BN 128
#define BK 32
#define LDSP 40   // padded LDS row pitch (ushort); 80B keeps b128 alignment, spreads banks

template <int MODE>
__global__ __launch_bounds__(256)
void gemm_bf16(const unsigned short* __restrict__ A, int lda,
               const unsigned short* __restrict__ Bw, int ldb,
               int M, int N, int K,
               const float* __restrict__ bias,
               const float* __restrict__ gates,
               float* __restrict__ outF,
               unsigned short* __restrict__ outH,
               int ldo,
               const float* __restrict__ xconv,
               const float* __restrict__ convW,
               const float* __restrict__ convB,
               int Lseq, int Dmod) {
  __shared__ unsigned short As[2 * BM * LDSP];   // 20 KB
  __shared__ unsigned short Bs[2 * BN * LDSP];   // 20 KB

  const int tid  = threadIdx.x;
  const int lane = tid & 31;
  const int wave = tid >> 5;      // 0..7
  const int wy = wave >> 2;       // 0..1  (M direction, 64 rows each)
  const int wx = wave & 3;        // 0..3  (N direction, 32 cols each)
  const int bm = blockIdx.y * BM;
  const int bn = blockIdx.x * BN;

  const int ldRow = tid >> 1;          // 0..127
  const int ldCol = (tid & 1) * 16;    // 0 or 16 (bf16 elems)
  const size_t aBase = (size_t)(bm + ldRow) * lda + ldCol;
  const size_t bBase = (size_t)(bn + ldRow) * ldb + ldCol;
  const int ldsOff = ldRow * LDSP + ldCol;

  const v8f vzero = {0.f, 0.f, 0.f, 0.f, 0.f, 0.f, 0.f, 0.f};
  v8f acc[4][2];
#pragma unroll
  for (int i = 0; i < 4; ++i)
#pragma unroll
    for (int j = 0; j < 2; ++j) acc[i][j] = vzero;

  // wave32 fragment addressing per CDNA5 ISA layouts
  const int rcSel  = lane & 15;                 // A row / B col within 16
  const int aKbase = (lane & 16) ? 8 : 0;       // A: interleaved 8-wide K halves
  const int bKbase = (lane & 16) ? 8 : 0;       // B: interleaved 8-wide K halves

  auto issue_stage = [&](int s, int buf) {
    const int k0 = s * BK;
    const unsigned short* ga = A  + aBase + k0;
    const unsigned short* gb = Bw + bBase + k0;
    unsigned short* la = As + buf * (BM * LDSP) + ldsOff;
    unsigned short* lb = Bs + buf * (BN * LDSP) + ldsOff;
    async_b128(la,     ga);
    async_b128(la + 8, ga + 8);
    async_b128(lb,     gb);
    async_b128(lb + 8, gb + 8);
  };

  auto compute_stage = [&](int buf) {
    const unsigned short* Ac = As + buf * (BM * LDSP);
    const unsigned short* Bc = Bs + buf * (BN * LDSP);
    FragAB af[4], bfr[2];
#pragma unroll
    for (int i = 0; i < 4; ++i) {
      const int row = wy * 64 + i * 16 + rcSel;
      af[i].q[0] = *(const u32x4*)&Ac[row * LDSP + aKbase];
      af[i].q[1] = *(const u32x4*)&Ac[row * LDSP + aKbase + 16];
    }
#pragma unroll
    for (int j = 0; j < 2; ++j) {
      const int col = wx * 32 + j * 16 + rcSel;
      bfr[j].q[0] = *(const u32x4*)&Bc[col * LDSP + bKbase];
      bfr[j].q[1] = *(const u32x4*)&Bc[col * LDSP + bKbase + 16];
    }
#pragma unroll
    for (int i = 0; i < 4; ++i)
#pragma unroll
      for (int j = 0; j < 2; ++j)
        acc[i][j] = __builtin_amdgcn_wmma_f32_16x16x32_bf16(
            false, af[i].m, false, bfr[j].m, (short)0, acc[i][j], false, false);
  };

  const int S = K / BK;   // >= 16 for all call sites
  issue_stage(0, 0);

  int s = 0;
  for (; s < S - 1; ++s) {           // steady state: branch-free body
    issue_stage(s + 1, (s + 1) & 1); // next tile DMA in flight under this tile's math
    wait_async_le<4>();              // oldest 4 (stage s) have landed; in-order done
    __syncthreads();
    compute_stage(s & 1);
    __syncthreads();                 // all waves done reading buf before its next DMA
  }
  // peeled last stage
  wait_async_le<0>();
  __syncthreads();
  compute_stage(s & 1);

  // ---- epilogue ----
  const int nLane = lane & 15;
  const int mHalf = (lane >> 4) * 8;
#pragma unroll
  for (int i = 0; i < 4; ++i) {
#pragma unroll
    for (int j = 0; j < 2; ++j) {
      const int gn = bn + wx * 32 + j * 16 + nLane;
      float w0 = 0.f, w1 = 0.f, w2 = 0.f, cbv = 0.f, bz = 0.f;
      if (MODE == 3) {
        w0 = convW[gn * 3 + 0]; w1 = convW[gn * 3 + 1]; w2 = convW[gn * 3 + 2];
        cbv = convB[gn];
      } else if (bias) {
        bz = bias[gn];
      }
#pragma unroll
      for (int e = 0; e < 8; ++e) {
        const int gm = bm + wy * 64 + i * 16 + mHalf + e;
        float v = acc[i][j][e];
        if (MODE == 0) {
          v += bz; v = v > 0.f ? v : 0.f;
          outH[(size_t)gm * ldo + gn] = f2bf(v);
        } else if (MODE == 1) {
          v += bz; v = 1.f / (1.f + __expf(-v));
          outF[(size_t)gm * ldo + gn] = v;
        } else if (MODE == 2) {
          v += bz; v *= gates[(size_t)gm * ldo + gn];
          outF[(size_t)gm * ldo + gn] = v;
        } else {
          const int b = gm / Lseq;
          const int l = gm - b * Lseq;
          const float* xr = xconv + (size_t)gm * Dmod + gn;
          float t = xr[0] * w1 + cbv;
          if (l > 0)         t += xr[-Dmod] * w0;
          if (l < Lseq - 1)  t += xr[ Dmod] * w2;
          outF[(size_t)gm * ldo + gn] = v + t;
        }
      }
    }
  }
}

// -------------------- segmented prefix scan over L --------------------
#define NSEG 8
#define SEGLEN 256   // L / NSEG

__global__ void scan_partial(const float* __restrict__ uG,
                             float* __restrict__ segsum, int Lseq, int R) {
  const int r = blockIdx.x * blockDim.x + threadIdx.x;
  const int s = blockIdx.y & (NSEG - 1);
  const int b = blockIdx.y >> 3;
  size_t base = ((size_t)b * Lseq + (size_t)s * SEGLEN) * R + r;
  float sum = 0.f;
  for (int i = 0; i < SEGLEN; ++i) sum += uG[base + (size_t)i * R];
  segsum[((size_t)(b * NSEG + s)) * R + r] = sum;
}

__global__ void scan_offsets(float* __restrict__ segsum, int R) {
  const int idx = blockIdx.x * blockDim.x + threadIdx.x;  // b*R + r
  const int b = idx / R;
  const int r = idx - b * R;
  float run = 0.f;
  for (int s = 0; s < NSEG; ++s) {
    size_t o = ((size_t)(b * NSEG + s)) * R + r;
    float t = segsum[o];
    segsum[o] = run;
    run += t;
  }
}

__global__ void scan_apply(const float* __restrict__ uG,
                           const float* __restrict__ vG,
                           const float* __restrict__ segsum,
                           unsigned short* __restrict__ goB, int Lseq, int R) {
  const int r = blockIdx.x * blockDim.x + threadIdx.x;
  const int s = blockIdx.y & (NSEG - 1);
  const int b = blockIdx.y >> 3;
  float run = segsum[((size_t)(b * NSEG + s)) * R + r];
  size_t base = ((size_t)b * Lseq + (size_t)s * SEGLEN) * R + r;
  for (int i = 0; i < SEGLEN; ++i) {
    size_t o = base + (size_t)i * R;
    run += uG[o];
    goB[o] = f2bf(run * vG[o]);
  }
}

// -------------------- launch --------------------
extern "C" void kernel_launch(void* const* d_in, const int* in_sizes, int n_in,
                              void* d_out, int out_size, void* d_ws, size_t ws_size,
                              hipStream_t stream) {
  (void)in_sizes; (void)n_in; (void)out_size; (void)ws_size;
  const float* x   = (const float*)d_in[0];
  const float* Wu  = (const float*)d_in[1];
  const float* bu  = (const float*)d_in[2];
  const float* Wv  = (const float*)d_in[3];
  const float* bv  = (const float*)d_in[4];
  const float* Wg1 = (const float*)d_in[5];
  const float* bg1 = (const float*)d_in[6];
  const float* Wg2 = (const float*)d_in[7];
  const float* bg2 = (const float*)d_in[8];
  const float* cw  = (const float*)d_in[9];
  const float* cb  = (const float*)d_in[10];
  float* out = (float*)d_out;

  const int Bb = 4, L = 2048, D = 2048, R = 2048, DH = 512;
  const int M = Bb * L;  // 8192

  // workspace layout (256B aligned). goB aliases xB (xB dead before scan_apply).
  char* ws = (char*)d_ws;
  size_t off = 0;
  auto alloc = [&](size_t bytes) -> void* {
    void* p = ws + off;
    off += (bytes + 255) & ~(size_t)255;
    return p;
  };
  unsigned short* xB   = (unsigned short*)alloc((size_t)M * D * 2);   // 33.5 MB
  unsigned short* WuB  = (unsigned short*)alloc((size_t)R * D * 2);   //  8.4 MB
  unsigned short* WvB  = (unsigned short*)alloc((size_t)R * D * 2);   //  8.4 MB
  unsigned short* Wg1B = (unsigned short*)alloc((size_t)DH * D * 2);  //  2.1 MB
  unsigned short* Wg2B = (unsigned short*)alloc((size_t)R * DH * 2);  //  2.1 MB
  unsigned short* hB   = (unsigned short*)alloc((size_t)M * DH * 2);  //  8.4 MB
  float* gatesF = (float*)alloc((size_t)M * R * 4);                   //   64 MB
  float* uG     = (float*)alloc((size_t)M * R * 4);                   //   64 MB
  float* vG     = (float*)alloc((size_t)M * R * 4);                   //   64 MB
  float* segsum = (float*)alloc((size_t)Bb * NSEG * R * 4);           // 0.25 MB
  unsigned short* goB = xB;  // alias: x bf16 no longer needed once scans run

  // 1) convert operands to bf16
  cvt_f32_bf16<<<4096, 256, 0, stream>>>(x,   xB,   (long long)M * D);
  cvt_f32_bf16<<<2048, 256, 0, stream>>>(Wu,  WuB,  (long long)R * D);
  cvt_f32_bf16<<<2048, 256, 0, stream>>>(Wv,  WvB,  (long long)R * D);
  cvt_f32_bf16<<<1024, 256, 0, stream>>>(Wg1, Wg1B, (long long)DH * D);
  cvt_f32_bf16<<<1024, 256, 0, stream>>>(Wg2, Wg2B, (long long)R * DH);

  // 2) h = relu(x @ Wg1^T + bg1)            (M x DH, K=D)
  gemm_bf16<0><<<dim3(DH / BN, M / BM), 256, 0, stream>>>(
      xB, D, Wg1B, D, M, DH, D, bg1, nullptr, nullptr, hB, DH,
      nullptr, nullptr, nullptr, L, D);

  // 3) gates = sigmoid(h @ Wg2^T + bg2)     (M x R, K=DH)
  gemm_bf16<1><<<dim3(R / BN, M / BM), 256, 0, stream>>>(
      hB, DH, Wg2B, DH, M, R, DH, bg2, nullptr, gatesF, nullptr, R,
      nullptr, nullptr, nullptr, L, D);

  // 4) u_gated = (x @ Wu^T + bu) * gates ;  v_gated = (x @ Wv^T + bv) * gates
  gemm_bf16<2><<<dim3(R / BN, M / BM), 256, 0, stream>>>(
      xB, D, WuB, D, M, R, D, bu, gatesF, uG, nullptr, R,
      nullptr, nullptr, nullptr, L, D);
  gemm_bf16<2><<<dim3(R / BN, M / BM), 256, 0, stream>>>(
      xB, D, WvB, D, M, R, D, bv, gatesF, vG, nullptr, R,
      nullptr, nullptr, nullptr, L, D);

  // 5) global_out = cumsum_L(u_gated) * v_gated   (segmented scan, bf16 out)
  scan_partial<<<dim3(R / 256, Bb * NSEG), 256, 0, stream>>>(uG, segsum, L, R);
  scan_offsets<<<(Bb * R) / 256, 256, 0, stream>>>(segsum, R);
  scan_apply<<<dim3(R / 256, Bb * NSEG), 256, 0, stream>>>(uG, vG, segsum, goB, L, R);

  // 6) out = depthwise_conv3(x) + global_out @ Wu^T   (M x D, K=R)
  gemm_bf16<3><<<dim3(D / BN, M / BM), 256, 0, stream>>>(
      goB, R, WuB, D, M, D, R, nullptr, nullptr, out, nullptr, D,
      x, cw, cb, L, D);
}
